// TrueE3EEAtomEncoder_69965017252204
// MI455X (gfx1250) — compile-verified, matrix operand order checked by reading, not run on hardware
//
#include <hip/hip_runtime.h>
#include <cstddef>

typedef __attribute__((ext_vector_type(16))) _Float16 v16h;
typedef __attribute__((ext_vector_type(8)))  float    v8f;

union Cv { unsigned u[8]; v16h h; };
union H2 { _Float16 h[2]; unsigned u; };

namespace ct {
constexpr int L1[11]   = {0,1,2,0,1,1,2,0,1,2,2};
constexpr int L2[11]   = {0,1,2,1,0,2,1,2,1,0,2};
constexpr int LOp[11]  = {0,0,0,1,1,1,1,2,2,2,2};
constexpr int UP[11]   = {64,32,16,64,32,32,16,64,32,16,16};
constexpr int WOFF[11] = {0,4096,6144,7168,9216,10240,11264,11776,12800,13312,13568};
constexpr int YOFF[11] = {0,64,96,112,304,400,496,544,864,1024,1104};
constexpr int CGOFF[11]= {0,1,10,35,44,53,98,143,168,213,238};
constexpr int MULS[3]  = {64,32,16};
constexpr int DIMS[3]  = {1,3,5};
constexpr int SEGS[3]  = {0,64,160};
constexpr int SHO[3]   = {0,1,4};
}

#define NBATCH 4
#define NATOM 40
#define BN 160
#define EE 6400
#define FEAT 240
#define WNUM 13824

// ---------------- Clebsch-Gordan init (tiny, once per launch) ----------------
__device__ double dfact_d(int n){ double r=1.0; while(n>1){ r*=n; n-=2; } return r; }
__device__ double mavg_d(int a,int b,int c){
  if ((a&1)||(b&1)||(c&1)) return 0.0;
  return dfact_d(a-1)*dfact_d(b-1)*dfact_d(c-1)/dfact_d(a+b+c+1);
}
__device__ int poly_get(int l,int i,double c[2],int e[2][3]){
  const double s3=1.7320508075688772, s5=2.23606797749979, s15=3.872983346207417;
  e[0][0]=e[0][1]=e[0][2]=e[1][0]=e[1][1]=e[1][2]=0; c[1]=0.0;
  if (l==0){ c[0]=1.0; return 1; }
  if (l==1){ c[0]=s3; e[0][i]=1; return 1; }
  switch(i){
    case 0: c[0]=s15; e[0][0]=1; e[0][1]=1; return 1;
    case 1: c[0]=s15; e[0][1]=1; e[0][2]=1; return 1;
    case 2: c[0]=1.5*s5; e[0][2]=2; c[1]=-0.5*s5; return 2;
    case 3: c[0]=s15; e[0][0]=1; e[0][2]=1; return 1;
    default: c[0]=0.5*s15; e[0][0]=2; c[1]=-0.5*s15; e[1][1]=2; return 2;
  }
}
__global__ void cg_init_kernel(float* cg){
  int p = threadIdx.x; if (p>=11) return;
  int l1=ct::L1[p], l2=ct::L2[p], l3=ct::LOp[p];
  int d1=2*l1+1, d2=2*l2+1, d3=2*l3+1;
  double G[125]; double ss=0.0;
  for (int i=0;i<d1;i++) for (int j=0;j<d2;j++) for (int k=0;k<d3;k++){
    double c1[2],c2[2],c3[2]; int e1[2][3],e2[2][3],e3[2][3];
    int n1=poly_get(l1,i,c1,e1), n2=poly_get(l2,j,c2,e2), n3=poly_get(l3,k,c3,e3);
    double s=0.0;
    for (int a=0;a<n1;a++) for (int b=0;b<n2;b++) for (int cc=0;cc<n3;cc++)
      s += c1[a]*c2[b]*c3[cc]*mavg_d(e1[a][0]+e2[b][0]+e3[cc][0],
                                     e1[a][1]+e2[b][1]+e3[cc][1],
                                     e1[a][2]+e2[b][2]+e3[cc][2]);
    G[(i*d2+j)*d3+k]=s; ss+=s*s;
  }
  double scale = sqrt((double)(2*l3+1)/ss);
  for (int idx=0; idx<d1*d2*d3; idx++) cg[ct::CGOFF[p]+idx] = (float)(G[idx]*scale);
}

// ---------------- wm3 -> f16 (col-major packed pairs along K) ----------------
__global__ void wm3_convert_kernel(const float* __restrict__ wm3, unsigned* __restrict__ wm3t){
  long idx = (long)blockIdx.x*256 + threadIdx.x;
  long total = 3L*WNUM*64;
  if (idx>=total) return;
  int t = (int)(idx/(WNUM*64)); int rem = (int)(idx%(WNUM*64));
  int col = rem/64, cp = rem%64;
  const float* base = wm3 + (long)t*128*WNUM;
  H2 cv; cv.h[0]=(_Float16)base[(2*cp)*WNUM+col]; cv.h[1]=(_Float16)base[(2*cp+1)*WNUM+col];
  wm3t[idx]=cv.u;
}

// ---------------- node scalar init ----------------
__global__ void node_init_kernel(const int* __restrict__ z, const float* __restrict__ pos,
                                 const unsigned char* __restrict__ mask,
                                 const float* __restrict__ z_emb, const float* __restrict__ w_in,
                                 float* __restrict__ x){
  int node = blockIdx.x; int b = node/NATOM, n = node%NATOM;
  __shared__ float si[81];
  int t = threadIdx.x; // 96 threads
  float maskf = mask[node] ? 1.f : 0.f;
  if (t<64) si[t] = z_emb[z[node]*64 + t];
  else if (t==64) si[64] = (n==0)?1.f:0.f;
  else if (t<81){
    int k=t-65;
    float dx=pos[node*3+0]-pos[(b*NATOM)*3+0];
    float dy=pos[node*3+1]-pos[(b*NATOM)*3+1];
    float dz=pos[node*3+2]-pos[(b*NATOM)*3+2];
    float sq=dx*dx+dy*dy+dz*dz;
    float r=(sq>1e-12f)?sqrtf(sq):0.f;
    float d=fminf(r,6.f)-0.4f*(float)k;
    si[t]=expf(-6.25f*d*d);
  }
  __syncthreads();
  if (t<64){
    float s=0.f;
    for (int i=0;i<81;i++) s += si[i]*w_in[i*64+t];
    x[node*FEAT+t]=s*(1.f/9.f)*maskf;
  }
  for (int f=64+t; f<FEAT; f+=96) x[node*FEAT+f]=0.f;
}

// ---------------- per-edge RBF / SH / mask ----------------
__global__ void edge_prep_kernel(const float* __restrict__ pos, const unsigned char* __restrict__ mask,
                                 float* __restrict__ erbf, float* __restrict__ esh, float* __restrict__ ef){
  int e = blockIdx.x*blockDim.x + threadIdx.x; if (e>=EE) return;
  int b=e/1600, r2=e%1600, i=r2/NATOM, j=r2%NATOM;
  int si=b*NATOM+i, sj=b*NATOM+j;
  float vx=pos[sj*3+0]-pos[si*3+0];
  float vy=pos[sj*3+1]-pos[si*3+1];
  float vz=pos[sj*3+2]-pos[si*3+2];
  float sq=vx*vx+vy*vy+vz*vz;
  ef[e]=(sq<=36.f && sq>1e-16f && mask[si] && mask[sj])?1.f:0.f;
  float elen=(sq>1e-12f)?sqrtf(sq):0.f;
  float inv=1.f/fmaxf(elen,1e-8f);
  float x=vx*inv, y=vy*inv, zc=vz*inv;
  float rm=fminf(elen,6.f);
  for (int k=0;k<16;k++){ float d=rm-0.4f*(float)k; erbf[e*16+k]=expf(-6.25f*d*d); }
  const float s3=1.7320508075688772f, s5=2.23606797749979f, s15=3.872983346207417f;
  float* sh=esh+e*9;
  sh[0]=1.f; sh[1]=s3*x; sh[2]=s3*y; sh[3]=s3*zc;
  sh[4]=s15*x*y; sh[5]=s15*y*zc; sh[6]=0.5f*s5*(3.f*zc*zc-1.f);
  sh[7]=s15*x*zc; sh[8]=0.5f*s15*(x*x-y*y);
}

__global__ void deg_kernel(const float* __restrict__ ef, float* __restrict__ deg){
  int node = blockIdx.x*64 + threadIdx.x; if (node>=BN) return;
  int b=node/NATOM, j=node%NATOM;
  float s=0.f;
  for (int i=0;i<NATOM;i++) s += ef[b*1600 + i*NATOM + j];
  deg[node]=fmaxf(s,1.f);
}

// ---------------- edge MLP (stages 1,2) -> f16-packed h2 ----------------
__global__ void edge_mlp_kernel(const float* __restrict__ erbf,
                                const float* __restrict__ wm1, const float* __restrict__ bm1,
                                const float* __restrict__ wm2, const float* __restrict__ bm2,
                                int t, unsigned* __restrict__ h2p){
  int e=blockIdx.x, j=threadIdx.x; // 128 threads
  __shared__ float rb[16]; __shared__ float h1[128]; __shared__ float h2[128];
  if (j<16) rb[j]=erbf[e*16+j];
  __syncthreads();
  const float* W1=wm1+t*16*128;
  float s=bm1[t*128+j];
  for (int k=0;k<16;k++) s += rb[k]*W1[k*128+j];
  h1[j]=s/(1.f+expf(-s));
  __syncthreads();
  const float* W2=wm2+t*128*128;
  float s2=bm2[t*128+j];
  for (int k=0;k<128;k++) s2 += h1[k]*W2[k*128+j];
  h2[j]=s2/(1.f+expf(-s2));
  __syncthreads();
  if (j<64){ H2 cv; cv.h[0]=(_Float16)h2[2*j]; cv.h[1]=(_Float16)h2[2*j+1]; h2p[e*64+j]=cv.u; }
}

// ---------------- fused WMMA GEMM + tensor-product epilogue ----------------
template<int LOx, int WT>
__device__ void tp_unit(int lane, int e0, const unsigned* a_lds, const _Float16* y_lds,
                        const float* ef_l, const unsigned* __restrict__ wm3t_t,
                        const float* __restrict__ bm3_t, float* __restrict__ mout){
  constexpr int K   = ct::DIMS[LOx];
  constexpr int Wlo = ct::MULS[LOx];
  constexpr int SEGB= ct::SEGS[LOx];
  const float ISF = (LOx==0)?0.09449111825230679f:(LOx==1)?0.08333333333333333f:0.08838834764831845f;
  int n=lane&15, hi=lane>>4, hi4=hi*4;
  Cv a_cv[4];
#pragma unroll
  for (int s=0;s<4;s++){
#pragma unroll
    for (int v=0;v<4;v++){
      a_cv[s].u[v]   = a_lds[n*64 + s*16 + hi4 + v];
      a_cv[s].u[4+v] = a_lds[n*64 + s*16 + 8 + hi4 + v];
    }
  }
  float macc[8*K];
#pragma unroll
  for (int q=0;q<8*K;q++) macc[q]=0.f;
#pragma unroll
  for (int p=0;p<11;p++){
    if (ct::LOp[p]!=LOx) continue;
    const int U = ct::UP[p];
    for (int u=0;u<U;u++){
      int col = ct::WOFF[p] + u*Wlo + WT*16 + n;
      const unsigned* bp = wm3t_t + (size_t)col*64;
      v8f acc;
#pragma unroll
      for (int q=0;q<8;q++) acc[q]=0.f;
#pragma unroll
      for (int s=0;s<4;s++){
        Cv bcv;
#pragma unroll
        for (int v=0;v<4;v++){ bcv.u[v]=bp[s*16+hi4+v]; bcv.u[4+v]=bp[s*16+8+hi4+v]; }
        acc = __builtin_amdgcn_wmma_f32_16x16x32_f16(false, a_cv[s].h, false, bcv.h,
                                                     (short)0, acc, false, false);
      }
      float bias = bm3_t[col];
#pragma unroll
      for (int v=0;v<8;v++){
        float wv = acc[v] + bias;
        const _Float16* yp = y_lds + (v+8*hi)*1184 + ct::YOFF[p] + u*K;
#pragma unroll
        for (int k=0;k<K;k++) macc[k*8+v] += (float)yp[k]*wv;
      }
    }
  }
#pragma unroll
  for (int v=0;v<8;v++){
    int r=v+8*hi; int e=e0+r; float sc=ISF*ef_l[r];
#pragma unroll
    for (int k=0;k<K;k++){
      int feat = SEGB + (WT*16+n)*K + k;
      mout[(size_t)e*FEAT + feat] = macc[k*8+v]*sc;
    }
  }
}

__global__ __launch_bounds__(256)
void fused_tp_kernel(const unsigned* __restrict__ h2p, const unsigned* __restrict__ wm3t_t,
                     const float* __restrict__ bm3_t, const float* __restrict__ cg,
                     const float* __restrict__ xcur, const float* __restrict__ esh,
                     const float* __restrict__ ef, float* __restrict__ mout){
  __shared__ unsigned a_lds[16*64];
  __shared__ _Float16 y_lds[16*1184];
  __shared__ float xsrc[16*FEAT];
  __shared__ float esh_l[16*9];
  __shared__ float ef_l[16];
  int tid=threadIdx.x; int e0=blockIdx.x*16;
  for (int idx=tid; idx<1024; idx+=256) a_lds[idx]=h2p[e0*64+idx];
  for (int idx=tid; idx<144; idx+=256) esh_l[idx]=esh[e0*9+idx];
  if (tid<16) ef_l[tid]=ef[e0+tid];
  for (int idx=tid; idx<16*FEAT; idx+=256){
    int r=idx/FEAT, c=idx%FEAT;
    int e=e0+r; int b=e/1600; int i=(e%1600)/NATOM;
    xsrc[idx]=xcur[(b*NATOM+i)*FEAT+c];
  }
  __syncthreads();
  // per-edge, per-path y[u,k] = sum_i xs[u,i] * (sum_j sh[j]*CG[i,j,k])
  for (int task=tid; task<176; task+=256){
    int r=task/11, p=task%11;
    int l1=ct::L1[p], l2=ct::L2[p], lo=ct::LOp[p];
    int d1=2*l1+1, d2=2*l2+1, K=ct::DIMS[lo];
    float P[25];
    const float* cgp = cg + ct::CGOFF[p];
    const float* shp = esh_l + r*9 + ct::SHO[l2];
    for (int i1=0;i1<d1;i1++) for (int k=0;k<K;k++){
      float s=0.f;
      for (int j=0;j<d2;j++) s += shp[j]*cgp[(i1*d2+j)*K + k];
      P[i1*K+k]=s;
    }
    int U=ct::UP[p];
    const float* xb = xsrc + r*FEAT + ct::SEGS[l1];
    _Float16* yo = y_lds + r*1184 + ct::YOFF[p];
    for (int u=0;u<U;u++) for (int k=0;k<K;k++){
      float s=0.f;
      for (int i1=0;i1<d1;i1++) s += xb[u*d1+i1]*P[i1*K+k];
      yo[u*K+k]=(_Float16)s;
    }
  }
  __syncthreads();
  int wid=tid>>5, lane=tid&31;
  switch(wid){
    case 0: tp_unit<0,0>(lane,e0,a_lds,y_lds,ef_l,wm3t_t,bm3_t,mout); break;
    case 1: tp_unit<0,1>(lane,e0,a_lds,y_lds,ef_l,wm3t_t,bm3_t,mout); break;
    case 2: tp_unit<0,2>(lane,e0,a_lds,y_lds,ef_l,wm3t_t,bm3_t,mout); break;
    case 3: tp_unit<0,3>(lane,e0,a_lds,y_lds,ef_l,wm3t_t,bm3_t,mout); break;
    case 4: tp_unit<1,0>(lane,e0,a_lds,y_lds,ef_l,wm3t_t,bm3_t,mout); break;
    case 5: tp_unit<1,1>(lane,e0,a_lds,y_lds,ef_l,wm3t_t,bm3_t,mout); break;
    case 6: tp_unit<2,0>(lane,e0,a_lds,y_lds,ef_l,wm3t_t,bm3_t,mout); break;
    default: break;
  }
}

// ---------------- deterministic per-dst aggregation ----------------
__global__ void agg_kernel(const float* __restrict__ m, const float* __restrict__ deg,
                           float* __restrict__ agg){
  int node=blockIdx.x; int f=threadIdx.x; if (f>=FEAT) return;
  int b=node/NATOM, j=node%NATOM;
  float s=0.f;
  for (int i=0;i<NATOM;i++) s += m[((size_t)(b*1600 + i*NATOM + j))*FEAT + f];
  agg[node*FEAT+f]=s/deg[node];
}

// ---------------- x = x + irlin_self(x) + irlin_out(agg) ----------------
__global__ void update_kernel(const float* __restrict__ xcur, const float* __restrict__ agg,
                              const float* __restrict__ ws0, const float* __restrict__ ws1,
                              const float* __restrict__ ws2, const float* __restrict__ wo0,
                              const float* __restrict__ wo1, const float* __restrict__ wo2,
                              int t, float* __restrict__ xnext){
  int node=blockIdx.x; int f=threadIdx.x;
  __shared__ float xr[FEAT]; __shared__ float ar[FEAT];
  for (int idx=f; idx<FEAT; idx+=blockDim.x){ xr[idx]=xcur[node*FEAT+idx]; ar[idx]=agg[node*FEAT+idx]; }
  __syncthreads();
  if (f<FEAT){
    int l=(f<64)?0:(f<160)?1:2;
    int base=ct::SEGS[l], M=ct::MULS[l], D=ct::DIMS[l];
    int local=f-base, w=local/D, d=local%D;
    const float* Ws=(l==0)?ws0+t*4096:(l==1)?ws1+t*1024:ws2+t*256;
    const float* Wo=(l==0)?wo0+t*4096:(l==1)?wo1+t*1024:wo2+t*256;
    float s1=0.f, s2=0.f;
    for (int u=0;u<M;u++){
      s1 += xr[base+u*D+d]*Ws[u*M+w];
      s2 += ar[base+u*D+d]*Wo[u*M+w];
    }
    float inv=rsqrtf((float)M);
    xnext[node*FEAT+f]=xr[f]+s1*inv+s2*inv;
  }
}

// ---------------- layernorm + mask ----------------
__global__ void ln_kernel(const float* __restrict__ x, const float* __restrict__ g,
                          const float* __restrict__ bvec, const unsigned char* __restrict__ mask,
                          float* __restrict__ out){
  int node=blockIdx.x; int tid=threadIdx.x;
  __shared__ float red[256]; __shared__ float red2[256];
  float s=0.f, s2=0.f;
  for (int f=tid; f<FEAT; f+=256){ float v=x[node*FEAT+f]; s+=v; s2+=v*v; }
  red[tid]=s; red2[tid]=s2; __syncthreads();
  for (int st=128; st>0; st>>=1){
    if (tid<st){ red[tid]+=red[tid+st]; red2[tid]+=red2[tid+st]; }
    __syncthreads();
  }
  float mu=red[0]/(float)FEAT;
  float var=red2[0]/(float)FEAT - mu*mu;
  float rstd=rsqrtf(var+1e-5f);
  float maskf=mask[node]?1.f:0.f;
  for (int f=tid; f<FEAT; f+=256){
    float v=x[node*FEAT+f];
    out[node*FEAT+f]=((v-mu)*rstd*g[f]+bvec[f])*maskf;
  }
}

// ---------------- launcher ----------------
extern "C" void kernel_launch(void* const* d_in, const int* in_sizes, int n_in,
                              void* d_out, int out_size, void* d_ws, size_t ws_size,
                              hipStream_t stream){
  (void)in_sizes; (void)n_in; (void)out_size; (void)ws_size;
  const int*   z     = (const int*)d_in[0];
  const float* pos   = (const float*)d_in[1];
  const unsigned char* mask = (const unsigned char*)d_in[2];
  const float* z_emb = (const float*)d_in[3];
  const float* w_in  = (const float*)d_in[4];
  const float* wm1   = (const float*)d_in[5];
  const float* bm1   = (const float*)d_in[6];
  const float* wm2   = (const float*)d_in[7];
  const float* bm2   = (const float*)d_in[8];
  const float* wm3   = (const float*)d_in[9];
  const float* bm3   = (const float*)d_in[10];
  const float* ws0   = (const float*)d_in[11];
  const float* ws1   = (const float*)d_in[12];
  const float* ws2   = (const float*)d_in[13];
  const float* wo0   = (const float*)d_in[14];
  const float* wo1   = (const float*)d_in[15];
  const float* wo2   = (const float*)d_in[16];
  const float* ln_g  = (const float*)d_in[17];
  const float* ln_b  = (const float*)d_in[18];

  char* base=(char*)d_ws; size_t off=0;
  auto alloc=[&](size_t bytes)->char*{ char* p=base+off; off=(off+bytes+255)&~(size_t)255; return p; };
  float*    cg   = (float*)   alloc(363*4);
  unsigned* wm3t = (unsigned*)alloc(3ull*WNUM*64*4);
  unsigned* h2p  = (unsigned*)alloc((size_t)EE*64*4);
  float*    xa   = (float*)   alloc((size_t)BN*FEAT*4);
  float*    xb   = (float*)   alloc((size_t)BN*FEAT*4);
  float*    agg  = (float*)   alloc((size_t)BN*FEAT*4);
  float*    m    = (float*)   alloc((size_t)EE*FEAT*4);
  float*    erbf = (float*)   alloc((size_t)EE*16*4);
  float*    esh  = (float*)   alloc((size_t)EE*9*4);
  float*    ef   = (float*)   alloc((size_t)EE*4);
  float*    deg  = (float*)   alloc((size_t)BN*4);

  cg_init_kernel<<<1,32,0,stream>>>(cg);
  long tot = 3L*WNUM*64;
  wm3_convert_kernel<<<(int)((tot+255)/256),256,0,stream>>>(wm3, wm3t);
  node_init_kernel<<<BN,96,0,stream>>>(z,pos,mask,z_emb,w_in,xa);
  edge_prep_kernel<<<(EE+127)/128,128,0,stream>>>(pos,mask,erbf,esh,ef);
  deg_kernel<<<(BN+63)/64,64,0,stream>>>(ef,deg);

  float* xc=xa; float* xn=xb;
  for (int t=0;t<3;t++){
    edge_mlp_kernel<<<EE,128,0,stream>>>(erbf,wm1,bm1,wm2,bm2,t,h2p);
    fused_tp_kernel<<<EE/16,256,0,stream>>>(h2p, wm3t+(size_t)t*WNUM*64, bm3+t*WNUM,
                                            cg, xc, esh, ef, m);
    agg_kernel<<<BN,256,0,stream>>>(m,deg,agg);
    update_kernel<<<BN,256,0,stream>>>(xc,agg,ws0,ws1,ws2,wo0,wo1,wo2,t,xn);
    float* tmp=xc; xc=xn; xn=tmp;
  }
  ln_kernel<<<BN,256,0,stream>>>(xc, ln_g, ln_b, mask, (float*)d_out);
}
